// GANClassfier_20916490732048
// MI455X (gfx1250) — matrix operand order repaired and checked by reading, not compiled, not run on hardware
//
#include <hip/hip_runtime.h>
#include <hip/hip_bf16.h>
#include <math.h>

typedef _Float16 h16;
typedef __attribute__((ext_vector_type(16))) _Float16 v16h;
typedef __attribute__((ext_vector_type(8)))  _Float16 v8h;
typedef __attribute__((ext_vector_type(8)))  float    v8f;

#define EPS 1e-5f

// ---------------------------------------------------------------------------
// Per-column (feature) batch stats -> folded BN scale/shift
//   scale[c] = g[c] * rsqrt(var[c]+eps);  shift[c] = b[c] - mean[c]*scale[c]
// ---------------------------------------------------------------------------
__global__ void colstats_kernel(const float* __restrict__ X, int Nr, int ld,
                                const float* __restrict__ g,
                                const float* __restrict__ b,
                                float* __restrict__ scale,
                                float* __restrict__ shift) {
    const int col = blockIdx.x;
    const int t = threadIdx.x;
    float s = 0.f, s2 = 0.f;
    for (int r = t; r < Nr; r += 256) {
        float v = X[(size_t)r * ld + col];
        s += v; s2 += v * v;
    }
    __shared__ float rs[256], rq[256];
    rs[t] = s; rq[t] = s2; __syncthreads();
    for (int off = 128; off > 0; off >>= 1) {
        if (t < off) { rs[t] += rs[t + off]; rq[t] += rq[t + off]; }
        __syncthreads();
    }
    if (t == 0) {
        float mean = rs[0] / (float)Nr;
        float var  = rq[0] / (float)Nr - mean * mean;
        float sc   = g[col] * rsqrtf(var + EPS);
        scale[col] = sc;
        shift[col] = b[col] - mean * sc;
    }
}

// ---------------------------------------------------------------------------
// Fold BN into weights: Wf16[h,k] = f16(W[h,k]*scale[k]);
//                       cf[h]     = c[h] + sum_k W[h,k]*shift[k]
// ---------------------------------------------------------------------------
__global__ void fold_kernel(const float* __restrict__ W, int K,
                            const float* __restrict__ scale,
                            const float* __restrict__ shift,
                            const float* __restrict__ c,
                            h16* __restrict__ Wf, float* __restrict__ cf) {
    const int h = blockIdx.x;
    const int t = threadIdx.x;
    const float* wr = W + (size_t)h * K;
    h16* wo = Wf + (size_t)h * K;
    float acc = 0.f;
    for (int k = t; k < K; k += 256) {
        float w = wr[k];
        wo[k] = (h16)(w * scale[k]);
        acc += w * shift[k];
    }
    __shared__ float red[256];
    red[t] = acc; __syncthreads();
    for (int off = 128; off > 0; off >>= 1) {
        if (t < off) red[t] += red[t + off];
        __syncthreads();
    }
    if (t == 0) cf[h] = (c ? c[h] : 0.f) + red[0];
}

// Plain f32 -> f16 convert
__global__ void cvt16_kernel(const float* __restrict__ in, h16* __restrict__ out, int n) {
    int i = blockIdx.x * blockDim.x + threadIdx.x;
    if (i < n) out[i] = (h16)in[i];
}

// ---------------------------------------------------------------------------
// WMMA helpers
// ---------------------------------------------------------------------------
// A 16x32 f16 fragment (ISA 7.12.2): lane holds row (lane&15);
// lanes 0-15 pack K 0-7 & 16-23, lanes 16-31 pack K 8-15 & 24-31.
// Caller pre-offsets the row pointer by hi*8 halves; we load [0..7] and [16..23].
__device__ __forceinline__ v16h load_afrag(const h16* __restrict__ p) {
    v8h lo  = *(const v8h*)p;
    v8h hi8 = *(const v8h*)(p + 16);
    v16h a;
#pragma unroll
    for (int e = 0; e < 8; ++e) { a[e] = lo[e]; a[e + 8] = hi8[e]; }
    return a;
}

__device__ __forceinline__ v8f wmma_f16(v16h a, v16h b, v8f c) {
    return __builtin_amdgcn_wmma_f32_16x16x32_f16(
        /*neg_a=*/false, a, /*neg_b=*/false, b,
        /*c_mod=*/(short)0, c, /*reuse_a=*/false, /*reuse_b=*/false);
}

__device__ __forceinline__ void store_tile(v8f c, int mbase, int ncol, int hi,
                                           const float* __restrict__ bias,
                                           float* __restrict__ outF,
                                           h16* __restrict__ outH,
                                           int ldc, int act) {
    const float bv = bias ? bias[ncol] : 0.f;
#pragma unroll
    for (int r = 0; r < 8; ++r) {
        int m = mbase + r + hi * 8;               // C/D layout: row = r + 8*(lane>=16)
        float v = c[r] + bv;
        if (act) v = 0.5f * v * (1.f + erff(v * 0.70710678118654752f));
        outF[(size_t)m * ldc + ncol] = v;
        if (outH) outH[(size_t)m * ldc + ncol] = (h16)v;
    }
}

// ---------------------------------------------------------------------------
// WMMA GEMM:  C[M,Nc] = act( A[M,K](f16) @ W[Nc,K](f16)^T + bias )
// 32x32 output tile per wave (2x2 sub-tiles, 4 accumulators): each k-step
// loads 2 A frags + 2 B frags and issues 4 independent WMMAs, halving
// bytes/WMMA vs a 16x16 tile and giving the scheduler ILP to hide latency.
// 4 waves/block tile along N. Requires M%32==0, Nc%128==0, K%32==0.
// act: 0 = identity, 1 = exact GELU. Writes f32 and (optionally) f16.
// ---------------------------------------------------------------------------
__global__ void wmma_gemm_kernel(const h16* __restrict__ A, int lda,
                                 const h16* __restrict__ W,
                                 const float* __restrict__ bias,
                                 float* __restrict__ outF,
                                 h16* __restrict__ outH, int ldc,
                                 int K, int act) {
    const int lane = threadIdx.x & 31;
    const int wv   = threadIdx.x >> 5;              // 0..3
    const int n0   = blockIdx.x * 128 + wv * 32;    // wave's first output column
    const int m0   = blockIdx.y * 32;               // wave's first output row

    const int lh = lane & 15;
    const int hi = lane >> 4;                       // 0: lanes 0-15, 1: lanes 16-31

    const h16* arow0 = A + (size_t)(m0 + lh) * lda + hi * 8;
    const h16* arow1 = arow0 + (size_t)16 * lda;
    // B 32x16 fragment = W^T: lane holds column; same K packing as A
    // (lanes 0-15: K 0-7 & 16-23, lanes 16-31: K 8-15 & 24-31).
    const h16* wrow0 = W + (size_t)(n0 + lh) * K + hi * 8;
    const h16* wrow1 = wrow0 + (size_t)16 * K;

    v8f c00 = {}, c01 = {}, c10 = {}, c11 = {};
    for (int k0 = 0; k0 < K; k0 += 32) {
        v16h a0 = load_afrag(arow0 + k0);
        v16h a1 = load_afrag(arow1 + k0);
        v16h b0 = load_afrag(wrow0 + k0);
        v16h b1 = load_afrag(wrow1 + k0);
        c00 = wmma_f16(a0, b0, c00);
        c01 = wmma_f16(a0, b1, c01);
        c10 = wmma_f16(a1, b0, c10);
        c11 = wmma_f16(a1, b1, c11);
    }

    const int nc0 = n0 + lh, nc1 = n0 + 16 + lh;
    store_tile(c00, m0,      nc0, hi, bias, outF, outH, ldc, act);
    store_tile(c01, m0,      nc1, hi, bias, outF, outH, ldc, act);
    store_tile(c10, m0 + 16, nc0, hi, bias, outF, outH, ldc, act);
    store_tile(c11, m0 + 16, nc1, hi, bias, outF, outH, ldc, act);
}

// ---------------------------------------------------------------------------
// Attention coefficients: a_src[i,h] = sum_d XH[i,h*64+d]*Wa[d]
//                         a_tgt[i,h] = sum_d XH[i,h*64+d]*Wa[64+d]
// ---------------------------------------------------------------------------
__global__ void attcoef_kernel(const float* __restrict__ XH,
                               const float* __restrict__ Wa,
                               float* __restrict__ a_src,
                               float* __restrict__ a_tgt) {
    const int i = blockIdx.x, t = threadIdx.x;   // 256 threads = 4 heads * 64 dims
    const int h = t >> 6, d = t & 63;
    __shared__ float rs[256], rt[256];
    float p = XH[(size_t)i * 256 + t];
    rs[t] = p * Wa[d];
    rt[t] = p * Wa[64 + d];
    __syncthreads();
    for (int off = 32; off > 0; off >>= 1) {
        if (d < off) { rs[t] += rs[t + off]; rt[t] += rt[t + off]; }
        __syncthreads();
    }
    if (d == 0) { a_src[i * 4 + h] = rs[t]; a_tgt[i * 4 + h] = rt[t]; }
}

// ---------------------------------------------------------------------------
// Per-node masked softmax attention + aggregation + residual + LayerNorm.
// One block per target node i; adjacency row is ~33 nonzeros of 4096.
// ---------------------------------------------------------------------------
__global__ void gat_kernel(const float* __restrict__ XH,
                           const float* __restrict__ a_src,
                           const float* __restrict__ a_tgt,
                           const unsigned char* __restrict__ adj,
                           const float* __restrict__ Xin,
                           const float* __restrict__ ln_g,
                           const float* __restrict__ ln_b,
                           float* __restrict__ XoutF,
                           h16* __restrict__ XoutH, int N) {
    const int i = blockIdx.x, t = threadIdx.x;   // 256 threads
    __shared__ float red[256];
    __shared__ float sacc[256];
    __shared__ float ssum[4];
    __shared__ float smax[4];
    __shared__ float smu, svar;

    const unsigned char* arow = adj + (size_t)i * N;
    // Speculative prefetch of this block's streaming rows (global_prefetch_b8)
    __builtin_prefetch(arow + t * 16, 0, 3);
    __builtin_prefetch(Xin + (size_t)i * 256 + t, 0, 3);

    float at[4];
#pragma unroll
    for (int h = 0; h < 4; ++h) at[h] = a_tgt[i * 4 + h];

    // Pass 1: per-head running max of masked leaky_relu scores
    float lmax[4] = {-1e30f, -1e30f, -1e30f, -1e30f};
    for (int j = t; j < N; j += 256) {
        if (arow[j]) {
#pragma unroll
            for (int h = 0; h < 4; ++h) {
                float e = at[h] + a_src[j * 4 + h];
                e = e > 0.f ? e : 0.2f * e;
                lmax[h] = fmaxf(lmax[h], e);
            }
        }
    }
#pragma unroll
    for (int h = 0; h < 4; ++h) {
        red[t] = lmax[h]; __syncthreads();
        for (int off = 128; off > 0; off >>= 1) {
            if (t < off) red[t] = fmaxf(red[t], red[t + off]);
            __syncthreads();
        }
        if (t == 0) smax[h] = red[0];
        __syncthreads();
    }

    sacc[t] = 0.f;
    if (t < 4) ssum[t] = 0.f;
    __syncthreads();

    // Pass 2: exp-weights + LDS-atomic aggregation of neighbor features
    for (int j = t; j < N; j += 256) {
        if (arow[j]) {
            const float* xr = XH + (size_t)j * 256;
#pragma unroll
            for (int h = 0; h < 4; ++h) {
                float e = at[h] + a_src[j * 4 + h];
                e = e > 0.f ? e : 0.2f * e;
                float w = expf(e - smax[h]);
                atomicAdd(&ssum[h], w);
                for (int d = 0; d < 64; ++d)
                    atomicAdd(&sacc[h * 64 + d], w * xr[h * 64 + d]);
            }
        }
    }
    __syncthreads();

    // residual + LayerNorm over the 256-dim feature
    const int h = t >> 6;
    float xv = Xin[(size_t)i * 256 + t] + sacc[t] / ssum[h];

    red[t] = xv; __syncthreads();
    for (int off = 128; off > 0; off >>= 1) {
        if (t < off) red[t] += red[t + off];
        __syncthreads();
    }
    if (t == 0) smu = red[0] * (1.f / 256.f);
    __syncthreads();
    float mu = smu;
    float dv = xv - mu;
    red[t] = dv * dv; __syncthreads();
    for (int off = 128; off > 0; off >>= 1) {
        if (t < off) red[t] += red[t + off];
        __syncthreads();
    }
    if (t == 0) svar = red[0] * (1.f / 256.f);
    __syncthreads();

    float o = dv * rsqrtf(svar + EPS) * ln_g[t] + ln_b[t];
    XoutF[(size_t)i * 256 + t] = o;
    XoutH[(size_t)i * 256 + t] = (h16)o;
}

// ---------------------------------------------------------------------------
// Gather + logits: out[r,c] = X[idx[r],:] . Wl[c,:] + cl[c]   (NCLS=40)
// ---------------------------------------------------------------------------
__global__ void cls_kernel(const float* __restrict__ X,
                           const int* __restrict__ idx,
                           const float* __restrict__ Wl,
                           const float* __restrict__ cl,
                           float* __restrict__ out) {
    __shared__ float xr[256];
    const int r = blockIdx.x, t = threadIdx.x;   // 64 threads
    const int node = idx[r];
    __builtin_prefetch(Wl + (size_t)t * 256, 0, 3);
    for (int k = t; k < 256; k += 64) xr[k] = X[(size_t)node * 256 + k];
    __syncthreads();
    if (t < 40) {
        const float* w = Wl + (size_t)t * 256;
        float acc = cl[t];
        for (int k = 0; k < 256; ++k) acc += xr[k] * w[k];
        out[(size_t)r * 40 + t] = acc;
    }
}

// ---------------------------------------------------------------------------
// Host orchestration
// ---------------------------------------------------------------------------
extern "C" void kernel_launch(void* const* d_in, const int* in_sizes, int n_in,
                              void* d_out, int out_size, void* d_ws, size_t ws_size,
                              hipStream_t stream) {
    (void)in_sizes; (void)n_in; (void)out_size; (void)ws_size;
    const int N = 4096, F = 512, HID = 1024, OUT = 256, NQ = 8192;

    const float* x      = (const float*)d_in[0];
    const float* pg1    = (const float*)d_in[1];
    const float* pb1    = (const float*)d_in[2];
    const float* pW1    = (const float*)d_in[3];
    const float* pc1    = (const float*)d_in[4];
    const float* pg2    = (const float*)d_in[5];
    const float* pb2    = (const float*)d_in[6];
    const float* pW2    = (const float*)d_in[7];
    const float* pc2    = (const float*)d_in[8];
    const float* Wp     = (const float*)d_in[9];
    const float* Wa     = (const float*)d_in[10];
    const float* ln_g   = (const float*)d_in[11];
    const float* ln_b   = (const float*)d_in[12];
    const float* qg1    = (const float*)d_in[13];
    const float* qb1    = (const float*)d_in[14];
    const float* qW1    = (const float*)d_in[15];
    const float* qc1    = (const float*)d_in[16];
    const float* qg2    = (const float*)d_in[17];
    const float* qb2    = (const float*)d_in[18];
    const float* qW2    = (const float*)d_in[19];
    const float* qc2    = (const float*)d_in[20];
    const float* Wl     = (const float*)d_in[21];
    const float* cl     = (const float*)d_in[22];
    const unsigned char* adj = (const unsigned char*)d_in[23];
    const int*   nidx   = (const int*)d_in[24];
    float* out = (float*)d_out;

    // Workspace layout (bytes); XH32 aliases Y32, X3_32 aliases x16 (disjoint lifetimes)
    char* ws = (char*)d_ws;
    float* Y32   = (float*)(ws + 0);                  // 16 MiB  [N,HID] f32
    h16*   Y16   = (h16*)  (ws + 16777216);           //  8 MiB  [N,HID] f16
    h16*   A16   = (h16*)  (ws + 25165824);           //  4 MiB  [N,F]   f16 (input)
    float* X32   = (float*)(ws + 29360128);           //  4 MiB  [N,OUT] f32
    h16*   X16   = (h16*)  (ws + 33554432);           //  2 MiB  [N,OUT] f16
    float* X2_32 = (float*)(ws + 35651584);           //  4 MiB  post-LN f32
    h16*   X2_16 = (h16*)  (ws + 39845888);           //  2 MiB  post-LN f16
    h16*   WF    = (h16*)  (ws + 41943040);           //  1 MiB  folded weights f16
    float* CF    = (float*)(ws + 42991616);           //  4 KiB  folded bias
    float* SCALE = (float*)(ws + 42995712);           //  4 KiB
    float* SHIFT = (float*)(ws + 42999808);           //  4 KiB
    float* ASRC  = (float*)(ws + 43003904);           // 64 KiB  [N,4]
    float* ATGT  = (float*)(ws + 43069440);           // 64 KiB  [N,4]
    float* XH32  = Y32;                               // alias: [N,OUT] xh f32
    float* X3_32 = (float*)A16;                       // alias: final FFN out f32

    // ---- preprocess FFN ----
    colstats_kernel<<<F, 256, 0, stream>>>(x, N, F, pg1, pb1, SCALE, SHIFT);
    fold_kernel<<<HID, 256, 0, stream>>>(pW1, F, SCALE, SHIFT, pc1, WF, CF);
    cvt16_kernel<<<(N * F + 255) / 256, 256, 0, stream>>>(x, A16, N * F);
    wmma_gemm_kernel<<<dim3(HID / 128, N / 32), 128, 0, stream>>>(
        A16, F, WF, CF, Y32, Y16, HID, F, /*gelu*/1);

    colstats_kernel<<<HID, 256, 0, stream>>>(Y32, N, HID, pg2, pb2, SCALE, SHIFT);
    fold_kernel<<<OUT, 256, 0, stream>>>(pW2, HID, SCALE, SHIFT, pc2, WF, CF);
    wmma_gemm_kernel<<<dim3(OUT / 128, N / 32), 128, 0, stream>>>(
        Y16, HID, WF, CF, X32, X16, OUT, HID, /*gelu*/1);

    // ---- GAT: xh = x @ Wp^T ----
    cvt16_kernel<<<(OUT * OUT + 255) / 256, 256, 0, stream>>>(Wp, WF, OUT * OUT);
    wmma_gemm_kernel<<<dim3(OUT / 128, N / 32), 128, 0, stream>>>(
        X16, OUT, WF, nullptr, XH32, nullptr, OUT, OUT, /*none*/0);

    attcoef_kernel<<<N, 256, 0, stream>>>(XH32, Wa, ASRC, ATGT);
    gat_kernel<<<N, 256, 0, stream>>>(XH32, ASRC, ATGT, adj, X32,
                                      ln_g, ln_b, X2_32, X2_16, N);

    // ---- postprocess FFN ----
    colstats_kernel<<<OUT, 256, 0, stream>>>(X2_32, N, OUT, qg1, qb1, SCALE, SHIFT);
    fold_kernel<<<HID, 256, 0, stream>>>(qW1, OUT, SCALE, SHIFT, qc1, WF, CF);
    wmma_gemm_kernel<<<dim3(HID / 128, N / 32), 128, 0, stream>>>(
        X2_16, OUT, WF, CF, Y32, Y16, HID, OUT, /*gelu*/1);

    colstats_kernel<<<HID, 256, 0, stream>>>(Y32, N, HID, qg2, qb2, SCALE, SHIFT);
    fold_kernel<<<OUT, 256, 0, stream>>>(qW2, HID, SCALE, SHIFT, qc2, WF, CF);
    wmma_gemm_kernel<<<dim3(OUT / 128, N / 32), 128, 0, stream>>>(
        Y16, HID, WF, CF, X3_32, nullptr, OUT, HID, /*gelu*/1);

    // ---- gather + logits ----
    cls_kernel<<<NQ, 64, 0, stream>>>(X3_32, nidx, Wl, cl, out);
}
